// GatedAttentionUnit_80247168959185
// MI455X (gfx1250) — compile-verified, matrix operand order checked by reading, not run on hardware
//
#include <hip/hip_runtime.h>
#include <hip/hip_bf16.h>
#include <math.h>

// ---------------------------------------------------------------------------
// Gated Attention Unit for MI455X (gfx1250, wave32, WMMA bf16).
// All matmuls -> v_wmma_f32_16x16x32_bf16. 256x128 block tiles, BK=64,
// 8 waves x (64x64) each. Double-buffered LDS fed by async global->LDS
// (GLOBAL_LOAD_ASYNC_TO_LDS_B128 + s_wait_asynccnt) when available.
// All B operands are [N][K] row-major (weights pre-transposed; v written
// transposed by its own epilogue) so every tile fill is contiguous 16B.
// P = relu^2(causal(q k^T / sqrt(I))) materialized bf16 (67 MB, fits L2).
// ---------------------------------------------------------------------------

typedef __attribute__((ext_vector_type(16))) __bf16 v16bf;
typedef __attribute__((ext_vector_type(8)))  __bf16 v8bf;
typedef __attribute__((ext_vector_type(8)))  float  v8f;
typedef int v4i __attribute__((ext_vector_type(4)));
typedef v4i __attribute__((address_space(1))) v4i_as1;  // global int4
typedef v4i __attribute__((address_space(3))) v4i_as3;  // LDS int4

#define BM 256
#define BN 128
#define BK 64
#define LDP 72  // padded LDS row (bf16) : 144B stride, 16B aligned

#if __has_builtin(__builtin_amdgcn_global_load_async_to_lds_b128) && \
    __has_builtin(__builtin_amdgcn_s_wait_asynccnt)
#define USE_ASYNC 1
#else
#define USE_ASYNC 0
#endif

union V16U { v16bf v; v8bf h[2]; };

__device__ __forceinline__ void cp16(const __bf16* g, __bf16* l) {
#if USE_ASYNC
    __builtin_amdgcn_global_load_async_to_lds_b128(
        (v4i_as1*)(uintptr_t)g, (v4i_as3*)(uint32_t)(uintptr_t)l, 0, 0);
#else
    *(uint4*)l = *(const uint4*)g;
#endif
}

__device__ __forceinline__ void tile_sync() {
#if USE_ASYNC
    __builtin_amdgcn_s_wait_asynccnt(0);
#endif
    __syncthreads();
}

// Copy ROWS x 64 bf16 tile (row-major, leading dim ld) into padded LDS rows.
template <int ROWS>
__device__ __forceinline__ void load_tile(const __bf16* __restrict__ src, long ld,
                                          __bf16 (*dst)[LDP], int tid) {
    for (int c = tid; c < ROWS * 8; c += 256) {
        int r = c >> 3, kc = (c & 7) << 3;  // 16B chunks
        cp16(src + (long)r * ld + kc, &dst[r][kc]);
    }
}

// CDNA5 16-bit A/B fragment: lane L holds row/col L%16;
// K-chunks [8h,8h+8) and [16+8h,16+8h+8) with h = L/16.
__device__ __forceinline__ v16bf load_frag(const __bf16* rowptr, int half) {
    V16U u;
    u.h[0] = *(const v8bf*)(rowptr + 8 * half);
    u.h[1] = *(const v8bf*)(rowptr + 16 + 8 * half);
    return u.v;
}

__device__ __forceinline__ float silu_f(float x) {
    return x * __builtin_amdgcn_rcpf(1.0f + __expf(-x));
}

// EPI: 0 silu->bf16 row-major (gate)     1 silu,q/k affine -> two bf16 outs
//      2 scale+causal+relu^2 -> bf16 (P) 3 multiply by gate -> bf16 (o)
//      4 +bias -> f32 (final)            5 silu -> bf16 TRANSPOSED (v^T)
// CMODE: 0 none, 1 causal tile-skip on cols (scores), 2 causal K-trunc (P@v)
template <int EPI, int CMODE>
__global__ __launch_bounds__(256) void gemm256(
    const __bf16* __restrict__ A, long ldA, long sA,
    const __bf16* __restrict__ Bmat, long ldB, long sB,
    const float* __restrict__ bias,
    void* __restrict__ out0, void* __restrict__ out1, long ldC, long sC,
    const __bf16* __restrict__ gate,
    const float* __restrict__ gq, const float* __restrict__ bq,
    const float* __restrict__ gk, const float* __restrict__ bk,
    int K, float scale)
{
    const int row0 = blockIdx.y * BM;
    const int col0 = blockIdx.x * BN;
    if (CMODE == 1 && col0 > row0 + BM - 1) return;  // fully above diagonal

    const int z = blockIdx.z;
    const __bf16* Ap = A + (long)z * sA + (long)row0 * ldA;
    const __bf16* Bp = Bmat + (long)z * sB + (long)col0 * ldB;
    const long outOff = (long)z * sC;

    __shared__ __align__(16) __bf16 As[2][BM][LDP];  // 72 KB
    __shared__ __align__(16) __bf16 Bs[2][BN][LDP];  // 36 KB

    const int tid  = threadIdx.x;
    const int lane = tid & 31;
    const int wave = tid >> 5;   // 8 waves
    const int wm   = wave >> 1;  // 0..3 : 64 rows each
    const int wn   = wave & 1;   // 0..1 : 64 cols each
    const int half = lane >> 4;
    const int lr   = lane & 15;

    v8f acc[4][4];
    #pragma unroll
    for (int i = 0; i < 4; i++)
        #pragma unroll
        for (int j = 0; j < 4; j++)
            #pragma unroll
            for (int r = 0; r < 8; r++) acc[i][j][r] = 0.0f;

    int kmax = K;
    if (CMODE == 2) {  // P@v: keys only up to end of this query tile
        int lim = row0 + BM;
        kmax = lim < K ? lim : K;
    }

    // prime buffer 0
    load_tile<BM>(Ap, ldA, As[0], tid);
    load_tile<BN>(Bp, ldB, Bs[0], tid);

    int ib = 0;
    for (int k0 = 0; k0 < kmax; k0 += BK, ib ^= 1) {
        tile_sync();  // buf ib ready (all waves' async fills done)
        if (k0 + BK < kmax) {  // prefetch next tile into other buffer
            load_tile<BM>(Ap + (k0 + BK), ldA, As[ib ^ 1], tid);
            load_tile<BN>(Bp + (k0 + BK), ldB, Bs[ib ^ 1], tid);
        }
        #pragma unroll
        for (int s = 0; s < 2; s++) {  // two K=32 wmma steps per tile
            v16bf af[4], bfr[4];
            #pragma unroll
            for (int i = 0; i < 4; i++)
                af[i] = load_frag(&As[ib][wm * 64 + i * 16 + lr][32 * s], half);
            #pragma unroll
            for (int j = 0; j < 4; j++)
                bfr[j] = load_frag(&Bs[ib][wn * 64 + j * 16 + lr][32 * s], half);
            #pragma unroll
            for (int i = 0; i < 4; i++)
                #pragma unroll
                for (int j = 0; j < 4; j++)
                    acc[i][j] = __builtin_amdgcn_wmma_f32_16x16x32_bf16(
                        false, af[i], false, bfr[j], (short)0, acc[i][j],
                        false, false);
        }
    }

    // --- epilogue: C/D layout -> row = r + 8*half, col = lane%16 ---
    #pragma unroll
    for (int i = 0; i < 4; i++) {
        #pragma unroll
        for (int j = 0; j < 4; j++) {
            const int col = col0 + wn * 64 + j * 16 + lr;
            if (EPI == 5) {
                // v^T store: 8 consecutive rows for fixed col -> one 16B store
                const int rowb = row0 + wm * 64 + i * 16 + 8 * half;
                union { uint4 u; __bf16 b[8]; } pk;
                #pragma unroll
                for (int r = 0; r < 8; r++)
                    pk.b[r] = (__bf16)silu_f(acc[i][j][r] + bias[col]);
                *(uint4*)&((__bf16*)out0)[(long)col * ldC + rowb] = pk.u;
            } else {
                #pragma unroll
                for (int r = 0; r < 8; r++) {
                    const int row = row0 + wm * 64 + i * 16 + r + 8 * half;
                    const long idx = outOff + (long)row * ldC + col;
                    float val = acc[i][j][r];
                    if (EPI == 0) {
                        ((__bf16*)out0)[idx] = (__bf16)silu_f(val + bias[col]);
                    } else if (EPI == 1) {
                        float y = silu_f(val + bias[col]);
                        ((__bf16*)out0)[idx] = (__bf16)(y * gq[col] + bq[col]);
                        ((__bf16*)out1)[idx] = (__bf16)(y * gk[col] + bk[col]);
                    } else if (EPI == 2) {
                        float s = val * scale;
                        if (col > row) s = 0.0f;  // causal mask
                        s = fmaxf(s, 0.0f);
                        ((__bf16*)out0)[idx] = (__bf16)(s * s);
                    } else if (EPI == 3) {
                        ((__bf16*)out0)[idx] = (__bf16)(val * (float)gate[idx]);
                    } else {
                        ((float*)out0)[idx] = val + bias[col];
                    }
                }
            }
        }
    }
}

__global__ void cvt_f32_bf16(const float* __restrict__ s, __bf16* __restrict__ d, long n) {
    long i = (long)blockIdx.x * blockDim.x + threadIdx.x;
    long stride = (long)gridDim.x * blockDim.x;
    for (; i < n; i += stride) d[i] = (__bf16)s[i];
}

// dst[n][k] = (bf16)src[k][n]; src is [K][N] f32 row-major. 32x32 LDS tiles.
__global__ __launch_bounds__(256) void cvt_transpose(const float* __restrict__ src,
                                                     __bf16* __restrict__ dst,
                                                     int K, int N) {
    __shared__ float t[32][33];
    const int k0 = blockIdx.y * 32, n0 = blockIdx.x * 32;
    const int tx = threadIdx.x & 31, ty = threadIdx.x >> 5;  // 8 rows/pass
    for (int r = ty; r < 32; r += 8)
        t[r][tx] = src[(long)(k0 + r) * N + (n0 + tx)];
    __syncthreads();
    for (int r = ty; r < 32; r += 8)
        dst[(long)(n0 + r) * K + (k0 + tx)] = (__bf16)t[tx][r];
}

extern "C" void kernel_launch(void* const* d_in, const int* in_sizes, int n_in,
                              void* d_out, int out_size, void* d_ws, size_t ws_size,
                              hipStream_t stream) {
    const int  Bb = 2, Nn = 4096, E = 1024, H = 128, I = 2048;
    const long MN = (long)Bb * Nn;  // 8192 fused batch*seq rows

    const float* x  = (const float*)d_in[0];
    const float* Wv = (const float*)d_in[1];
    const float* bv = (const float*)d_in[2];
    const float* Wg = (const float*)d_in[3];
    const float* bg = (const float*)d_in[4];
    const float* Wi = (const float*)d_in[5];
    const float* bi = (const float*)d_in[6];
    const float* gq = (const float*)d_in[7];
    const float* bq = (const float*)d_in[8];
    const float* gk = (const float*)d_in[9];
    const float* bk = (const float*)d_in[10];
    const float* Wo = (const float*)d_in[11];
    const float* bo = (const float*)d_in[12];

    char* w = (char*)d_ws;
    auto alloc = [&](long bytes) -> char* {
        char* p = w;
        w += (bytes + 255) & ~255L;
        return p;
    };
    __bf16* xb   = (__bf16*)alloc(MN * E * 2);        // x, bf16 [MN][E]
    __bf16* WvbT = (__bf16*)alloc((long)E * I * 2);   // [I][E]
    __bf16* WgbT = (__bf16*)alloc((long)E * I * 2);   // [I][E]
    __bf16* WibT = (__bf16*)alloc((long)E * H * 2);   // [H][E]
    __bf16* WobT = (__bf16*)alloc((long)I * E * 2);   // [E][I]
    __bf16* vT   = (__bf16*)alloc(MN * I * 2);        // v transposed [I][MN]
    __bf16* gB   = (__bf16*)alloc(MN * I * 2);        // gate [MN][I]
    __bf16* qB   = (__bf16*)alloc(MN * H * 2);        // [MN][H]
    __bf16* kB   = (__bf16*)alloc(MN * H * 2);        // [MN][H]
    __bf16* P    = (__bf16*)alloc((long)Bb * Nn * Nn * 2);  // scores
    __bf16* oB   = (__bf16*)alloc(MN * I * 2);        // [MN][I]

    cvt_f32_bf16<<<1024, 256, 0, stream>>>(x, xb, MN * E);
    cvt_transpose<<<dim3(I / 32, E / 32), 256, 0, stream>>>(Wv, WvbT, E, I);
    cvt_transpose<<<dim3(I / 32, E / 32), 256, 0, stream>>>(Wg, WgbT, E, I);
    cvt_transpose<<<dim3(H / 32, E / 32), 256, 0, stream>>>(Wi, WibT, E, H);
    cvt_transpose<<<dim3(E / 32, I / 32), 256, 0, stream>>>(Wo, WobT, I, E);

    // v^T = silu(x @ Wv + bv)^T   [I][MN]
    gemm256<5, 0><<<dim3(I / BN, MN / BM, 1), 256, 0, stream>>>(
        xb, E, 0, WvbT, E, 0, bv, vT, nullptr, MN, 0,
        nullptr, nullptr, nullptr, nullptr, nullptr, E, 1.0f);
    // gate = silu(x @ Wg + bg)    [MN][I]
    gemm256<0, 0><<<dim3(I / BN, MN / BM, 1), 256, 0, stream>>>(
        xb, E, 0, WgbT, E, 0, bg, gB, nullptr, I, 0,
        nullptr, nullptr, nullptr, nullptr, nullptr, E, 1.0f);
    // x_ = silu(x @ Wi + bi); q = x_*gq+bq; k = x_*gk+bk
    gemm256<1, 0><<<dim3(H / BN, MN / BM, 1), 256, 0, stream>>>(
        xb, E, 0, WibT, E, 0, bi, qB, kB, H, 0,
        nullptr, gq, bq, gk, bk, E, 1.0f);
    // P = relu^2(causal(q k^T / sqrt(I)))  per batch, lower-triangular tiles
    const float scl = 0.022097086912079608f;  // 1/sqrt(2048)
    gemm256<2, 1><<<dim3(Nn / BN, Nn / BM, Bb), 256, 0, stream>>>(
        qB, H, (long)Nn * H, kB, H, (long)Nn * H, nullptr,
        P, nullptr, Nn, (long)Nn * Nn,
        nullptr, nullptr, nullptr, nullptr, nullptr, H, scl);
    // o = (P @ v) * gate ; B operand = v^T, batch offset = z*Nn along K
    gemm256<3, 2><<<dim3(I / BN, Nn / BM, Bb), 256, 0, stream>>>(
        P, Nn, (long)Nn * Nn, vT, MN, (long)Nn, nullptr,
        oB, nullptr, I, (long)Nn * I,
        gB, nullptr, nullptr, nullptr, nullptr, Nn, 1.0f);
    // out = o @ Wo + bo  (fp32)
    gemm256<4, 0><<<dim3(E / BN, MN / BM, 1), 256, 0, stream>>>(
        oB, I, 0, WobT, I, 0, bo, d_out, nullptr, E, 0,
        nullptr, nullptr, nullptr, nullptr, nullptr, I, 1.0f);
}
